// Attention_11433202942207
// MI455X (gfx1250) — matrix operand order repaired and checked by reading, not compiled, not run on hardware
//
#include <hip/hip_runtime.h>

// ---------------------------------------------------------------------------
// Types / WMMA helpers (gfx1250, wave32)
// ---------------------------------------------------------------------------
typedef __bf16 bf16;
typedef __attribute__((ext_vector_type(16))) __bf16 v16bf;
typedef __attribute__((ext_vector_type(8)))  float  v8f;

union FragU { uint4 q[2]; v16bf v; };

static __device__ __forceinline__ v8f vzero() {
  v8f z;
#pragma unroll
  for (int i = 0; i < 8; ++i) z[i] = 0.0f;
  return z;
}

static __device__ __forceinline__ bf16 f2bf(float x) {
  unsigned u = __builtin_bit_cast(unsigned, x);
  unsigned r = u + 0x7FFFu + ((u >> 16) & 1u);           // round-to-nearest-even
  unsigned short s = (unsigned short)(r >> 16);
  return __builtin_bit_cast(bf16, s);
}

// A fragment (16x32 MxK bf16): rowp points at this lane's row, k-offset 0 of
// the 32-wide K chunk.  16 contiguous K per lane: lanes 0-15 hold K 0..15,
// lanes 16-31 hold K 16..31 (matches load_bt).
static __device__ __forceinline__ v16bf load_a_row(const bf16* rowp) {
  int kb = ((threadIdx.x & 31) >> 4) * 16;
  FragU f;
  f.q[0] = *(const uint4*)(rowp + kb);
  f.q[1] = *(const uint4*)(rowp + kb + 8);
  return f.v;
}
static __device__ __forceinline__ v16bf load_a(const bf16* A, int lda) {
  return load_a_row(A + (size_t)(threadIdx.x & 15) * lda);
}

// B fragment (32x16 KxN bf16) sourced from an [N,K] row-major matrix W
// (i.e. B[k][n] = W[n][k]).  lane&15 = column, 16 contiguous K per lane:
// lanes 0-15 K=0..15, lanes 16-31 K=16..31.
static __device__ __forceinline__ v16bf load_bt(const bf16* W, int ldw) {
  int lane = threadIdx.x & 31;
  const bf16* p = W + (size_t)(lane & 15) * ldw + ((lane >> 4) * 16);
  FragU f;
  f.q[0] = *(const uint4*)p;
  f.q[1] = *(const uint4*)(p + 8);
  return f.v;
}

static __device__ __forceinline__ v8f wmma_bf16(v16bf a, v16bf b, v8f c) {
  return __builtin_amdgcn_wmma_f32_16x16x32_bf16(false, a, false, b,
                                                 (short)0, c, false, false);
}

// Problem constants
static constexpr int Bc   = 8;
static constexpr int Nc   = 4096;
static constexpr int Cc   = 512;
static constexpr int Hh   = 8;
static constexpr int Dd   = 64;
static constexpr int N2c  = 1024;   // (64/2)*(64/2)

// ---------------------------------------------------------------------------
// Dense GEMM core: 16 rows x 64 cols per wave, K=512, double-buffered so the
// next step's 10 b128 loads are in flight across the current 4 WMMAs.
// Wn0 = W + n0*512 ([N,K] row-major weight slice).
// ---------------------------------------------------------------------------
static __device__ __forceinline__ void gemm_core(const bf16* Arow, const bf16* Wn0,
                                                 v8f acc[4]) {
  v16bf a = load_a_row(Arow);
  v16bf b[4];
#pragma unroll
  for (int t = 0; t < 4; ++t) b[t] = load_bt(Wn0 + (size_t)t * 16 * Cc, Cc);
#pragma unroll
  for (int k = 0; k < Cc; k += 32) {
    const int k2 = (k + 32 < Cc) ? k + 32 : k;   // clamp on last step (dead data)
    v16bf an = load_a_row(Arow + k2);
    v16bf bn[4];
#pragma unroll
    for (int t = 0; t < 4; ++t) bn[t] = load_bt(Wn0 + (size_t)t * 16 * Cc + k2, Cc);
#pragma unroll
    for (int t = 0; t < 4; ++t) acc[t] = wmma_bf16(a, b[t], acc[t]);
    a = an;
#pragma unroll
    for (int t = 0; t < 4; ++t) b[t] = bn[t];
  }
}

// ---------------------------------------------------------------------------
// Prep kernels: fp32 -> bf16 conversions / weight repack
// ---------------------------------------------------------------------------
__global__ void pack_bf16_kernel(const float* __restrict__ in,
                                 bf16* __restrict__ out, int n) {
  int i = blockIdx.x * blockDim.x + threadIdx.x;
  if (i < n) out[i] = f2bf(in[i]);
}

// sr_w [Cout,Cin,2,2] -> Wsr [Cout, 2048] with K = (ky*2+kx)*512 + cin
__global__ void pack_sr_kernel(const float* __restrict__ sr_w,
                               bf16* __restrict__ out) {
  int i = blockIdx.x * blockDim.x + threadIdx.x;
  if (i >= Cc * 2048) return;
  int n   = i >> 11;
  int k   = i & 2047;
  int pix = k >> 9;
  int cin = k & 511;
  int ky = pix >> 1, kx = pix & 1;
  out[i] = f2bf(sr_w[((size_t)(n * Cc + cin) * 2 + ky) * 2 + kx]);
}

// ---------------------------------------------------------------------------
// GEMM: q = (x @ q_w^T + q_b) * d^-0.5  -> q_bf [B,h,N,d] bf16
// block = 256 threads = 8 waves; block tile = 16 rows x 512 cols
// ---------------------------------------------------------------------------
__global__ __launch_bounds__(256)
void gemm_q_kernel(const bf16* __restrict__ X, const bf16* __restrict__ W,
                   const float* __restrict__ bias, bf16* __restrict__ Q) {
  const int m0   = blockIdx.x * 16;
  const int wave = threadIdx.x >> 5;
  const int lane = threadIdx.x & 31;
  const int n0   = wave * 64;
  const bf16* Arow = X + (size_t)(m0 + (lane & 15)) * Cc;

  v8f acc[4];
#pragma unroll
  for (int t = 0; t < 4; ++t) acc[t] = vzero();
  gemm_core(Arow, W + (size_t)n0 * Cc, acc);

  const int colq = lane & 15, rbase = (lane >> 4) * 8;
#pragma unroll
  for (int t = 0; t < 4; ++t) {
    int c  = n0 + 16 * t + colq;
    int hh = c >> 6, dd = c & 63;
    float bv = bias[c];
#pragma unroll
    for (int r = 0; r < 8; ++r) {
      int m = m0 + rbase + r;
      int b = m >> 12, nr = m & 4095;
      Q[(((size_t)(b * Hh + hh)) * Nc + nr) * Dd + dd] =
          f2bf((acc[t][r] + bv) * 0.125f);   // fold softmax scale d^-0.5
    }
  }
}

// ---------------------------------------------------------------------------
// SR conv (gather GEMM, K=2048) + bias + LayerNorm -> x_sr_ln [8192,512] bf16
// ---------------------------------------------------------------------------
__global__ __launch_bounds__(256)
void gemm_sr_kernel(const bf16* __restrict__ Xb, const bf16* __restrict__ Wsr,
                    const float* __restrict__ sr_b,
                    const float* __restrict__ ln_g, const float* __restrict__ ln_b,
                    bf16* __restrict__ Out) {
  __shared__ float S[16][Cc + 4];
  __shared__ float R1[16][16], R2[16][16];

  const int m0   = blockIdx.x * 16;
  const int wave = threadIdx.x >> 5;
  const int lane = threadIdx.x & 31;
  const int n0   = wave * 64;

  // gather geometry for this lane's row
  const int m   = m0 + (lane & 15);
  const int b   = m >> 10;
  const int rem = m & 1023;
  const int ry  = rem >> 5, cx = rem & 31;

  auto arow_ptr = [&](int k) -> const bf16* {
    int pix = k >> 9;
    int ky = pix >> 1, kx = pix & 1;
    int c0 = k & 511;
    return Xb + ((size_t)(b * Nc + (2 * ry + ky) * 64 + (2 * cx + kx)) * Cc + c0);
  };

  v8f acc[4];
#pragma unroll
  for (int t = 0; t < 4; ++t) acc[t] = vzero();

  // double-buffered K loop (K = 2048)
  v16bf a = load_a_row(arow_ptr(0));
  v16bf bb[4];
#pragma unroll
  for (int t = 0; t < 4; ++t)
    bb[t] = load_bt(Wsr + (size_t)(n0 + 16 * t) * 2048, 2048);
  for (int k = 0; k < 2048; k += 32) {
    const int k2 = (k + 32 < 2048) ? k + 32 : k;
    v16bf an = load_a_row(arow_ptr(k2));
    v16bf bn[4];
#pragma unroll
    for (int t = 0; t < 4; ++t)
      bn[t] = load_bt(Wsr + (size_t)(n0 + 16 * t) * 2048 + k2, 2048);
#pragma unroll
    for (int t = 0; t < 4; ++t) acc[t] = wmma_bf16(a, bb[t], acc[t]);
    a = an;
#pragma unroll
    for (int t = 0; t < 4; ++t) bb[t] = bn[t];
  }

  const int colq = lane & 15, rbase = (lane >> 4) * 8;
#pragma unroll
  for (int t = 0; t < 4; ++t) {
    int c = n0 + 16 * t + colq;
    float bv = sr_b[c];
#pragma unroll
    for (int r = 0; r < 8; ++r) S[rbase + r][c] = acc[t][r] + bv;
  }
  __syncthreads();

  // LayerNorm over the 512-wide row strip
  const int row = threadIdx.x >> 4, seg = threadIdx.x & 15;
  float s1 = 0.0f, s2 = 0.0f;
#pragma unroll 8
  for (int j = 0; j < 32; ++j) {
    float v = S[row][seg * 32 + j];
    s1 += v; s2 += v * v;
  }
  R1[row][seg] = s1; R2[row][seg] = s2;
  __syncthreads();

  float mu = 0.0f, m2 = 0.0f;
#pragma unroll
  for (int j = 0; j < 16; ++j) { mu += R1[row][j]; m2 += R2[row][j]; }
  mu *= (1.0f / 512.0f);
  m2 = m2 * (1.0f / 512.0f) - mu * mu;
  float rstd = rsqrtf(m2 + 1e-5f);

  const int gm = m0 + row;
#pragma unroll 8
  for (int j = 0; j < 32; ++j) {
    int c = seg * 32 + j;
    float v = (S[row][c] - mu) * rstd * ln_g[c] + ln_b[c];
    Out[(size_t)gm * Cc + c] = f2bf(v);
  }
}

// ---------------------------------------------------------------------------
// kv = x_sr_ln @ kv_w^T + kv_b  ->  K natural [B,h,N2,d], V transposed [B,h,d,N2]
// grid (512, 2)
// ---------------------------------------------------------------------------
__global__ __launch_bounds__(256)
void gemm_kv_kernel(const bf16* __restrict__ X, const bf16* __restrict__ W,
                    const float* __restrict__ bias,
                    bf16* __restrict__ Kn, bf16* __restrict__ VT) {
  const int m0   = blockIdx.x * 16;
  const int wave = threadIdx.x >> 5;
  const int lane = threadIdx.x & 31;
  const int n0   = blockIdx.y * 512 + wave * 64;
  const bf16* Arow = X + (size_t)(m0 + (lane & 15)) * Cc;

  v8f acc[4];
#pragma unroll
  for (int t = 0; t < 4; ++t) acc[t] = vzero();
  gemm_core(Arow, W + (size_t)n0 * Cc, acc);

  const int colq = lane & 15, rbase = (lane >> 4) * 8;
#pragma unroll
  for (int t = 0; t < 4; ++t) {
    int c2 = n0 + 16 * t + colq;
    float bv = bias[c2];
    if (c2 < 512) {                       // k output (wave-uniform branch)
      int hh = c2 >> 6, dd = c2 & 63;
#pragma unroll
      for (int r = 0; r < 8; ++r) {
        int m = m0 + rbase + r;
        int b = m >> 10, n2 = m & 1023;
        Kn[(((size_t)(b * Hh + hh)) * N2c + n2) * Dd + dd] = f2bf(acc[t][r] + bv);
      }
    } else {                              // v output, stored transposed [d,N2]
      int cc = c2 - 512;
      int hh = cc >> 6, dd = cc & 63;
#pragma unroll
      for (int r = 0; r < 8; ++r) {
        int m = m0 + rbase + r;
        int b = m >> 10, n2 = m & 1023;
        VT[(((size_t)(b * Hh + hh)) * Dd + dd) * N2c + n2] = f2bf(acc[t][r] + bv);
      }
    }
  }
}

// ---------------------------------------------------------------------------
// Fused flash attention: per wave, one 16-row q-tile; stream K/V in 32-key
// chunks with online softmax.  grid (32, 64): x = 128-row q block, y = (b,h)
// Pipelining: V frags issued before the S WMMAs (stay in flight across
// softmax); next chunk's K frags issued right after the S WMMAs.
// ---------------------------------------------------------------------------
__global__ __launch_bounds__(256)
void attn_kernel(const bf16* __restrict__ Q, const bf16* __restrict__ Kn,
                 const bf16* __restrict__ VT, bf16* __restrict__ Out) {
  __shared__ bf16 Pbuf[8][16][40];        // per-wave P staging (16B-aligned rows)

  const int bh   = blockIdx.y;
  const int wave = threadIdx.x >> 5;
  const int lane = threadIdx.x & 31;
  const int q0   = blockIdx.x * 128 + wave * 16;

  const bf16* Qb = Q  + ((size_t)bh * Nc + q0) * Dd;
  const bf16* Kb = Kn + (size_t)bh * N2c * Dd;
  const bf16* Vb = VT + (size_t)bh * Dd * N2c;
  bf16 (*P)[40]  = Pbuf[wave];

  // q A-fragments (K-dim = 64 -> two 32-chunks), q already scaled by d^-0.5
  const bf16* qrow = Qb + (size_t)(lane & 15) * Dd;
  const v16bf qa0 = load_a_row(qrow);
  const v16bf qa1 = load_a_row(qrow + 32);

  v8f o[4];
#pragma unroll
  for (int t = 0; t < 4; ++t) o[t] = vzero();
  float mrow[8], lrow[8];
#pragma unroll
  for (int r = 0; r < 8; ++r) { mrow[r] = -1e30f; lrow[r] = 0.0f; }

  const int colq = lane & 15, rbase = (lane >> 4) * 8;

  // K fragments for the first chunk
  v16bf kf[4];
  kf[0] = load_bt(Kb,               Dd);
  kf[1] = load_bt(Kb + 32,          Dd);
  kf[2] = load_bt(Kb + 16 * Dd,      Dd);
  kf[3] = load_bt(Kb + 16 * Dd + 32, Dd);

  for (int kc = 0; kc < N2c; kc += 32) {
    // V fragments for this chunk, issued early to overlap S wmmas + softmax
    v16bf vf[4];
#pragma unroll
    for (int t = 0; t < 4; ++t)
      vf[t] = load_bt(Vb + (size_t)(t * 16) * N2c + kc, N2c);

    // S = q . k^T for 16 q rows x 32 keys (two 16-key n-tiles, K=64)
    v8f s0 = vzero(), s1 = vzero();
    s0 = wmma_bf16(qa0, kf[0], s0);
    s0 = wmma_bf16(qa1, kf[1], s0);
    s1 = wmma_bf16(qa0, kf[2], s1);
    s1 = wmma_bf16(qa1, kf[3], s1);

    // prefetch next chunk's K fragments (clamped on last iteration)
    {
      const int knc = (kc + 32 < N2c) ? kc + 32 : kc;
      const bf16* kp = Kb + (size_t)knc * Dd;
      kf[0] = load_bt(kp,               Dd);
      kf[1] = load_bt(kp + 32,          Dd);
      kf[2] = load_bt(kp + 16 * Dd,      Dd);
      kf[3] = load_bt(kp + 16 * Dd + 32, Dd);
    }

    // online softmax (row reductions across the 16-lane column groups)
#pragma unroll
    for (int r = 0; r < 8; ++r) {
      float mx = fmaxf(s0[r], s1[r]);
      mx = fmaxf(mx, __shfl_xor(mx, 1, 32));
      mx = fmaxf(mx, __shfl_xor(mx, 2, 32));
      mx = fmaxf(mx, __shfl_xor(mx, 4, 32));
      mx = fmaxf(mx, __shfl_xor(mx, 8, 32));
      float mnew  = fmaxf(mrow[r], mx);
      float alpha = __expf(mrow[r] - mnew);
      mrow[r] = mnew;
      float p0 = __expf(s0[r] - mnew);
      float p1 = __expf(s1[r] - mnew);
      float rs = p0 + p1;
      rs += __shfl_xor(rs, 1, 32);
      rs += __shfl_xor(rs, 2, 32);
      rs += __shfl_xor(rs, 4, 32);
      rs += __shfl_xor(rs, 8, 32);
      lrow[r] = lrow[r] * alpha + rs;
      s0[r] = p0; s1[r] = p1;
      o[0][r] *= alpha; o[1][r] *= alpha; o[2][r] *= alpha; o[3][r] *= alpha;
    }

    // stage P (C-layout -> A-layout) through per-wave LDS; DS ops from the
    // same wave are in-order, so no barrier is required.
#pragma unroll
    for (int r = 0; r < 8; ++r) {
      P[rbase + r][colq]      = f2bf(s0[r]);
      P[rbase + r][16 + colq] = f2bf(s1[r]);
    }
    v16bf pa = load_a(&P[0][0], 40);

    // O += P @ V  (V^T is [d, N2] -> [N,K]-form B source)
#pragma unroll
    for (int t = 0; t < 4; ++t) o[t] = wmma_bf16(pa, vf[t], o[t]);
  }

  // normalize and scatter to [B, N, C] bf16
  const int b = bh >> 3, hh = bh & 7;
#pragma unroll
  for (int r = 0; r < 8; ++r) {
    float inv = 1.0f / lrow[r];
    int qi = q0 + rbase + r;
#pragma unroll
    for (int t = 0; t < 4; ++t) {
      int c = hh * 64 + t * 16 + colq;
      Out[((size_t)b * Nc + qi) * Cc + c] = f2bf(o[t][r] * inv);
    }
  }
}

// ---------------------------------------------------------------------------
// Final projection: out = attn_out @ proj_w^T + proj_b   (fp32 output)
// ---------------------------------------------------------------------------
__global__ __launch_bounds__(256)
void gemm_proj_kernel(const bf16* __restrict__ X, const bf16* __restrict__ W,
                      const float* __restrict__ bias, float* __restrict__ Out) {
  const int m0   = blockIdx.x * 16;
  const int wave = threadIdx.x >> 5;
  const int lane = threadIdx.x & 31;
  const int n0   = wave * 64;
  const bf16* Arow = X + (size_t)(m0 + (lane & 15)) * Cc;

  v8f acc[4];
#pragma unroll
  for (int t = 0; t < 4; ++t) acc[t] = vzero();
  gemm_core(Arow, W + (size_t)n0 * Cc, acc);

  const int colq = lane & 15, rbase = (lane >> 4) * 8;
#pragma unroll
  for (int t = 0; t < 4; ++t) {
    int c = n0 + 16 * t + colq;
    float bv = bias[c];
#pragma unroll
    for (int r = 0; r < 8; ++r)
      Out[(size_t)(m0 + rbase + r) * Cc + c] = acc[t][r] + bv;
  }
}

// ---------------------------------------------------------------------------
// Launch
// ---------------------------------------------------------------------------
extern "C" void kernel_launch(void* const* d_in, const int* in_sizes, int n_in,
                              void* d_out, int out_size, void* d_ws, size_t ws_size,
                              hipStream_t stream) {
  const float* x      = (const float*)d_in[0];
  const float* q_w    = (const float*)d_in[1];
  const float* q_b    = (const float*)d_in[2];
  const float* kv_w   = (const float*)d_in[3];
  const float* kv_b   = (const float*)d_in[4];
  const float* sr_w   = (const float*)d_in[5];
  const float* sr_b   = (const float*)d_in[6];
  const float* ln_g   = (const float*)d_in[7];
  const float* ln_b   = (const float*)d_in[8];
  const float* proj_w = (const float*)d_in[9];
  const float* proj_b = (const float*)d_in[10];
  float* out = (float*)d_out;

  char* ws = (char*)d_ws;
  size_t off = 0;
  auto alloc = [&](size_t bytes) { void* p = ws + off; off += (bytes + 255) & ~(size_t)255; return p; };

  const size_t BN  = (size_t)Bc * Nc;        // 32768
  const size_t BN2 = (size_t)Bc * N2c;       // 8192

  bf16* x_bf     = (bf16*)alloc(BN * Cc * 2);        // later reused as attn_out
  bf16* q_bf     = (bf16*)alloc(BN * Cc * 2);
  bf16* x_sr_ln  = (bf16*)alloc(BN2 * Cc * 2);
  bf16* Kn       = (bf16*)alloc(BN2 * Cc * 2);
  bf16* VT       = (bf16*)alloc(BN2 * Cc * 2);
  bf16* Wq       = (bf16*)alloc((size_t)Cc * Cc * 2);
  bf16* Wkv      = (bf16*)alloc((size_t)2 * Cc * Cc * 2);
  bf16* Wsr      = (bf16*)alloc((size_t)Cc * 2048 * 2);
  bf16* Wproj    = (bf16*)alloc((size_t)Cc * Cc * 2);
  bf16* attn_out = x_bf;   // alias: x_bf's last reader is gemm_sr/gemm_q

  // 1. conversions / packs
  {
    int n = (int)(BN * Cc);
    pack_bf16_kernel<<<(n + 255) / 256, 256, 0, stream>>>(x, x_bf, n);
  }
  pack_bf16_kernel<<<(Cc * Cc + 255) / 256, 256, 0, stream>>>(q_w, Wq, Cc * Cc);
  pack_bf16_kernel<<<(2 * Cc * Cc + 255) / 256, 256, 0, stream>>>(kv_w, Wkv, 2 * Cc * Cc);
  pack_bf16_kernel<<<(Cc * Cc + 255) / 256, 256, 0, stream>>>(proj_w, Wproj, Cc * Cc);
  pack_sr_kernel<<<(Cc * 2048 + 255) / 256, 256, 0, stream>>>(sr_w, Wsr);

  // 2. q projection (scale folded in)
  gemm_q_kernel<<<(int)(BN / 16), 256, 0, stream>>>(x_bf, Wq, q_b, q_bf);

  // 3. SR conv + LayerNorm
  gemm_sr_kernel<<<(int)(BN2 / 16), 256, 0, stream>>>(x_bf, Wsr, sr_b, ln_g, ln_b, x_sr_ln);

  // 4. kv projection -> K [B,h,N2,d], V^T [B,h,d,N2]
  gemm_kv_kernel<<<dim3((unsigned)(BN2 / 16), 2), 256, 0, stream>>>(x_sr_ln, Wkv, kv_b, Kn, VT);

  // 5. fused flash attention (overwrites x_bf as attn_out)
  attn_kernel<<<dim3(Nc / 128, Bc * Hh), 256, 0, stream>>>(q_bf, Kn, VT, attn_out);

  // 6. output projection -> fp32 d_out
  gemm_proj_kernel<<<(int)(BN / 16), 256, 0, stream>>>(attn_out, Wproj, proj_b, out);

  (void)in_sizes; (void)n_in; (void)out_size; (void)ws_size;
}
